// GlycanGearNet_55645596287225
// MI455X (gfx1250) — matrix operand order repaired and checked by reading, not compiled, not run on hardware
//
#include <hip/hip_runtime.h>

#define NN 50000
#define EE 1600000
#define RR 7
#define DD 128
#define GG 64
#define KREL 896          // R*D
#define KTOT 1024         // R*D + D

typedef __attribute__((ext_vector_type(16))) __bf16 v16bf;
typedef __attribute__((ext_vector_type(8)))  float  v8f;
typedef __attribute__((ext_vector_type(4)))  float  f4;

// ---------------- zero (float4 granularity) ----------------
__global__ void k_zero(float* __restrict__ p, int n4) {
  int i = blockIdx.x * blockDim.x + threadIdx.x;
  if (i < n4) ((float4*)p)[i] = make_float4(0.f, 0.f, 0.f, 0.f);
}

// ---------------- x = emb[unit_type] ----------------
__global__ void k_init_x(const int* __restrict__ ut, const float* __restrict__ emb,
                         float* __restrict__ x) {
  int t = blockIdx.x * blockDim.x + threadIdx.x;   // N*32 threads, wave per node
  int n = t >> 5, l = t & 31;
  if (n >= NN) return;
  int u = ut[n];
  float4 v = *(const float4*)(emb + (size_t)u * DD + l * 4);
  *(float4*)(x + (size_t)n * DD + l * 4) = v;
}

// ---------------- edge scatter: upd[(no*R+rel)] += x[ni]*ew ----------------
__global__ void k_scatter(const float* __restrict__ x, const int* __restrict__ ni,
                          const int* __restrict__ no, const int* __restrict__ rel,
                          const float* __restrict__ ew, float* __restrict__ upd) {
  int t = blockIdx.x * blockDim.x + threadIdx.x;   // E*32 threads, wave per edge
  int e = t >> 5, l = t & 31;
  if (e >= EE) return;
  int a = ni[e];
  int b = no[e];
  int r = rel[e];
  float w = ew[e];
  float4 v = *(const float4*)(x + (size_t)a * DD + l * 4);
  float* d = upd + ((size_t)b * RR + r) * DD + l * 4;
  unsafeAtomicAdd(d + 0, v.x * w);
  unsafeAtomicAdd(d + 1, v.y * w);
  unsafeAtomicAdd(d + 2, v.z * w);
  unsafeAtomicAdd(d + 3, v.w * w);
}

// ---------------- pack [W_rel ; W_loop] (1024x128 f32) into bf16 B-fragments ---
// Wp index = ((s*8 + coltile)*32 + lane)*16 ; lane: col = lane&15, khalf = lane>>4
// element i (0..15): K = s*32 + khalf*16 + i   (contiguous k order, same as the A fill in k_gemm)
__global__ void k_pack_w(const float* __restrict__ Wrel, const float* __restrict__ Wloop,
                         __bf16* __restrict__ Wp) {
  int t = blockIdx.x * blockDim.x + threadIdx.x;   // 32*8*32 = 8192
  if (t >= 32 * 8 * 32) return;
  int lane = t & 31;
  int ct   = (t >> 5) & 7;
  int s    = t >> 8;
  int col  = ct * 16 + (lane & 15);
  int kh   = lane >> 4;
  __bf16* dst = Wp + (size_t)t * 16;
#pragma unroll
  for (int i = 0; i < 16; ++i) {
    int k = s * 32 + kh * 16 + i;
    float f = (k < KREL) ? Wrel[(size_t)k * DD + col]
                         : Wloop[(size_t)(k - KREL) * DD + col];
    dst[i] = (__bf16)f;
  }
}

// ---------------- GEMM: xout = relu([upd|x] @ Wp + b_rel + b_loop) ------------
// Block = 256 threads = 8 waves; 32-row node tile per block (two 16-row subtiles
// -> two accumulators per wave, so each B fragment feeds 2 WMMAs). Wave w owns
// column tile w (16 of 128 cols). K = 1024 -> 32 steps of v_wmma_f32_16x16x32_bf16.
__global__ void __launch_bounds__(256)
k_gemm(const float* __restrict__ upd, const float* __restrict__ x,
       const __bf16* __restrict__ Wp, const float* __restrict__ brel,
       const float* __restrict__ bloop, float* __restrict__ xout) {
  // packed A fragments: [mt 0..1][step 0..31][lane 0..31][16 bf16] = 64 KB
  __shared__ __align__(32) __bf16 Asmem[2 * 32 * 32 * 16];

  const int tid  = threadIdx.x;
  const int tile = blockIdx.x;   // 32-row tile

  // Cooperative fill: item = (mt, s, l); contiguous k order (same as Wp):
  // row = l&15, h = l>>4; elems 0..15 -> K = s*32+16h+i.
  // Source row is concat [upd row (896) | x row (128)]; NT loads keep L2 clean
  // (upd is 179 MB streamed exactly once per layer).
#pragma unroll
  for (int j = 0; j < 8; ++j) {
    int item = tid + 256 * j;            // 0..2047
    int l  = item & 31;
    int s  = (item >> 5) & 31;
    int mt = item >> 10;
    int row = l & 15, h = l >> 4;
    int node = tile * 32 + mt * 16 + row;
    v16bf frag = {};
    if (node < NN) {
#pragma unroll
      for (int half = 0; half < 2; ++half) {
        int k0 = s * 32 + 16 * h + 8 * half;   // 8-aligned; 896 is 8-aligned
        const float* src = (k0 < KREL) ? (upd + (size_t)node * KREL + k0)
                                       : (x + (size_t)node * DD + (k0 - KREL));
        f4 p0 = __builtin_nontemporal_load((const f4*)src);
        f4 p1 = __builtin_nontemporal_load(((const f4*)src) + 1);
#pragma unroll
        for (int i = 0; i < 4; ++i) {
          frag[half * 8 + i]     = (__bf16)p0[i];
          frag[half * 8 + 4 + i] = (__bf16)p1[i];
        }
      }
    }
    *(v16bf*)(Asmem + (size_t)item * 16) = frag;
  }
  __syncthreads();

  const int w    = tid >> 5;    // column tile 0..7
  const int lane = tid & 31;
  v8f acc0 = {}, acc1 = {};
  const v16bf* a0fr = (const v16bf*)Asmem;                    // rows tile*32 + 0..15
  const v16bf* a1fr = (const v16bf*)(Asmem + 32 * 32 * 16);   // rows tile*32 + 16..31
  const v16bf* bfr  = (const v16bf*)Wp;
#pragma unroll 4
  for (int s = 0; s < 32; ++s) {
    v16bf b  = bfr[(s * 8 + w) * 32 + lane];
    v16bf a0 = a0fr[s * 32 + lane];
    v16bf a1 = a1fr[s * 32 + lane];
    acc0 = __builtin_amdgcn_wmma_f32_16x16x32_bf16(
        false, a0, false, b, (short)0, acc0, false, false);
    acc1 = __builtin_amdgcn_wmma_f32_16x16x32_bf16(
        false, a1, false, b, (short)0, acc1, false, false);
  }

  // Epilogue. C/D layout: VGPR v -> row v (lanes 0-15) / row v+8 (lanes 16-31).
  // NN % 16 == 0, so each 16-row subtile is uniformly valid or invalid.
  int col = w * 16 + (lane & 15);
  int h = lane >> 4;
  float bias = brel[col] + bloop[col];
  int r0 = tile * 32 + 8 * h;
  if (tile * 32 + 16 <= NN) {
#pragma unroll
    for (int v = 0; v < 8; ++v) {
      float val = acc0[v] + bias;
      xout[(size_t)(r0 + v) * DD + col] = val > 0.f ? val : 0.f;
    }
  }
  if (tile * 32 + 32 <= NN) {
#pragma unroll
    for (int v = 0; v < 8; ++v) {
      float val = acc1[v] + bias;
      xout[(size_t)(r0 + 16 + v) * DD + col] = val > 0.f ? val : 0.f;
    }
  }
}

// ---------------- graph pooling: gout[n2g[n]] += x[n] ----------------
__global__ void k_pool(const float* __restrict__ xfin, const int* __restrict__ n2g,
                       float* __restrict__ gout) {
  int t = blockIdx.x * blockDim.x + threadIdx.x;   // N*32 threads, wave per node
  int n = t >> 5, l = t & 31;
  if (n >= NN) return;
  int g = n2g[n];
  float4 v = *(const float4*)(xfin + (size_t)n * DD + l * 4);
  float* d = gout + (size_t)g * DD + l * 4;
  unsafeAtomicAdd(d + 0, v.x);
  unsafeAtomicAdd(d + 1, v.y);
  unsafeAtomicAdd(d + 2, v.z);
  unsafeAtomicAdd(d + 3, v.w);
}

extern "C" void kernel_launch(void* const* d_in, const int* in_sizes, int n_in,
                              void* d_out, int out_size, void* d_ws, size_t ws_size,
                              hipStream_t stream) {
  const int*   ut    = (const int*)  d_in[0];
  const int*   ni    = (const int*)  d_in[1];
  const int*   no    = (const int*)  d_in[2];
  const int*   rel   = (const int*)  d_in[3];
  const float* ew    = (const float*)d_in[4];
  const int*   n2g   = (const int*)  d_in[5];
  const float* emb   = (const float*)d_in[6];
  const float* Wrel  = (const float*)d_in[7];
  const float* brel  = (const float*)d_in[8];
  const float* Wloop = (const float*)d_in[9];
  const float* bloop = (const float*)d_in[10];
  float* out = (float*)d_out;

  // workspace layout
  float*  upd = (float*)d_ws;                          // N*R*D   f32 (179.2 MB)
  float*  xA  = upd + (size_t)NN * RR * DD;            // N*D     f32 (25.6 MB)
  float*  xB  = xA + (size_t)NN * DD;                  // N*D     f32 (25.6 MB)
  __bf16* Wp  = (__bf16*)(xB + (size_t)NN * DD);       // 32*8*32*16 bf16 (256 KB)

  // x0 = emb[unit_type]; zero pooled-output region
  k_init_x<<<(NN * 32 + 255) / 256, 256, 0, stream>>>(ut, emb, xA);
  k_zero<<<((GG * DD / 4) + 255) / 256, 256, 0, stream>>>(out, GG * DD / 4);

  const float* xc = xA;
  for (int layer = 0; layer < 3; ++layer) {
    float* xn = (layer == 0) ? xB : (layer == 1) ? xA : (out + GG * DD);
    int n4 = NN * RR * DD / 4;
    k_zero<<<(n4 + 255) / 256, 256, 0, stream>>>(upd, n4);
    k_scatter<<<(EE * 32 + 255) / 256, 256, 0, stream>>>(xc, ni, no, rel, ew, upd);
    k_pack_w<<<(8192 + 255) / 256, 256, 0, stream>>>(
        Wrel + (size_t)layer * KREL * DD, Wloop + (size_t)layer * DD * DD, Wp);
    k_gemm<<<(NN + 31) / 32, 256, 0, stream>>>(
        upd, xc, Wp, brel + layer * DD, bloop + layer * DD, xn);
    xc = xn;
  }
  k_pool<<<(NN * 32 + 255) / 256, 256, 0, stream>>>(xc, n2g, out);
}
